// EncoderCS_14508399526596
// MI455X (gfx1250) — compile-verified, matrix operand order checked by reading, not run on hardware
//
#include <hip/hip_runtime.h>
#include <hip/hip_bf16.h>

#define BB   32
#define SS   1024
#define HHH  512
#define EMBD 512
#define MMM  (BB * SS)   // 32768 records

typedef __attribute__((ext_vector_type(16))) __bf16 v16bf;
typedef __attribute__((ext_vector_type(8)))  float  v8f;
typedef __attribute__((ext_vector_type(4)))  unsigned int u32x4;

union FragU { struct { u32x4 lo, hi; } u; v16bf v; };

__device__ __forceinline__ unsigned short f2bf(float f) {
  unsigned int u = __float_as_uint(f);
  u += 0x7FFFu + ((u >> 16) & 1u);   // round-to-nearest-even
  return (unsigned short)(u >> 16);
}

// Load one bf16 A/B fragment (16 lanes x 16 elems) for v_wmma_*_16x16x32_bf16.
// p points at (row, k0 + kbase) in a row-major bf16 matrix; the lane's 16
// elements are the contiguous runs [p, p+8) and [p+16, p+24).
__device__ __forceinline__ v16bf load_frag(const unsigned short* p) {
  FragU f;
  f.u.lo = *(const u32x4*)(p);
  f.u.hi = *(const u32x4*)(p + 16);
  return f.v;
}

// ---------------------------------------------------------------- prep ----
__global__ __launch_bounds__(256) void embed_gather_kernel(
    const int* __restrict__ types, const int* __restrict__ ents,
    const int* __restrict__ vals,  const int* __restrict__ ha,
    const float* __restrict__ Etp, const float* __restrict__ Eent,
    const float* __restrict__ Eval, const float* __restrict__ Eha,
    unsigned short* __restrict__ emb) {
  int r  = blockIdx.x;
  int ty = types[r], en = ents[r], va = vals[r], hw = ha[r];
  #pragma unroll
  for (int i = 0; i < 2; ++i) {
    int c = threadIdx.x + i * 256;
    float v;
    if      (c < 128) v = Etp [ty * 128 + c];
    else if (c < 256) v = Eent[en * 128 + (c - 128)];
    else if (c < 384) v = Eval[va * 128 + (c - 256)];
    else              v = Eha [hw * 128 + (c - 384)];
    emb[(size_t)r * EMBD + c] = f2bf(v);
  }
}

__global__ __launch_bounds__(256) void transpose_wenc_kernel(
    const float* __restrict__ W, unsigned short* __restrict__ WT) {
  int idx = blockIdx.x * 256 + threadIdx.x;        // 512*512
  int n = idx >> 9, k = idx & 511;
  WT[(size_t)n * 512 + k] = f2bf(W[(size_t)k * 512 + n]);
}

__global__ __launch_bounds__(256) void transpose_wg_kernel(
    const float* __restrict__ W, unsigned short* __restrict__ WT) {
  int idx = blockIdx.x * 256 + threadIdx.x;        // 512*1024
  int n = idx >> 10, k = idx & 1023;
  WT[(size_t)n * 1024 + k] = f2bf(W[(size_t)k * 512 + n]);
}

// ---------------------------------------------------- encode: h = relu ----
// h[M,H] = relu(emb[M,EMB] @ W_enc[EMB,H] + b).  One block per 16-row tile,
// 8 waves each own a 16x64 slab of N.
__global__ __launch_bounds__(256) void gemm_encode_kernel(
    const unsigned short* __restrict__ emb, const unsigned short* __restrict__ wT,
    const float* __restrict__ bias,
    float* __restrict__ h, unsigned short* __restrict__ hb,
    unsigned short* __restrict__ hT) {
  int lane  = threadIdx.x & 31;
  int wave  = threadIdx.x >> 5;
  int mbase = blockIdx.x * 16;
  int nbase = wave * 64;
  int arow  = mbase + (lane & 15);
  int kb    = (lane >> 4) << 3;
  v8f acc[4] = {};
  for (int k0 = 0; k0 < EMBD; k0 += 32) {
    v16bf a = load_frag(emb + (size_t)arow * EMBD + k0 + kb);
    #pragma unroll
    for (int j = 0; j < 4; ++j) {
      int brow = nbase + j * 16 + (lane & 15);
      v16bf b = load_frag(wT + (size_t)brow * EMBD + k0 + kb);
      acc[j] = __builtin_amdgcn_wmma_f32_16x16x32_bf16(
          false, a, false, b, (short)0, acc[j], false, false);
    }
  }
  int half = lane >> 4;
  #pragma unroll
  for (int j = 0; j < 4; ++j) {
    int n  = nbase + j * 16 + (lane & 15);
    float bn = bias[n];
    #pragma unroll
    for (int r = 0; r < 8; ++r) {
      int m = mbase + r + half * 8;
      float v = fmaxf(acc[j][r] + bn, 0.0f);
      size_t idx = (size_t)m * HHH + n;
      unsigned short bv = f2bf(v);
      h[idx]  = v;
      hb[idx] = bv;
      int bat = m >> 10, s = m & 1023;
      hT[(size_t)bat * HHH * SS + (size_t)n * SS + s] = bv;
    }
  }
}

// ------------------- fused: scores = h@h^T, self-mask, softmax -> attn ----
// One block per (batch, 16-query tile).  Each of the 8 waves computes a
// 16x128 score slab (8 WMMA accumulators), then row max/sum are reduced
// in-wave via shfl_xor (rows live in fixed 16-lane groups) and cross-wave
// via LDS; bf16 attention probabilities are written directly.
__global__ __launch_bounds__(256) void scores_softmax_kernel(
    const unsigned short* __restrict__ hb, unsigned short* __restrict__ attn) {
  int lane  = threadIdx.x & 31;
  int wave  = threadIdx.x >> 5;
  int bat   = blockIdx.y;
  int mbase = blockIdx.x * 16;
  int nbase = wave * 128;
  const unsigned short* hA = hb + (size_t)bat * SS * HHH;
  int arow = mbase + (lane & 15);
  int kb   = (lane >> 4) << 3;
  v8f acc[8] = {};
  for (int k0 = 0; k0 < HHH; k0 += 32) {
    v16bf a = load_frag(hA + (size_t)arow * HHH + k0 + kb);
    #pragma unroll
    for (int j = 0; j < 8; ++j) {
      int brow = nbase + j * 16 + (lane & 15);
      v16bf b = load_frag(hA + (size_t)brow * HHH + k0 + kb);
      acc[j] = __builtin_amdgcn_wmma_f32_16x16x32_bf16(
          false, a, false, b, (short)0, acc[j], false, false);
    }
  }
  int half = lane >> 4;
  int ncol = lane & 15;

  // self-mask: exclude j == t
  #pragma unroll
  for (int j = 0; j < 8; ++j) {
    int n = nbase + j * 16 + ncol;
    #pragma unroll
    for (int r = 0; r < 8; ++r) {
      if ((mbase + r + half * 8) == n) acc[j][r] = -1e9f;
    }
  }

  __shared__ float red_mx[16][8];
  __shared__ float red_sm[16][8];

  // per-row max over this wave's 128 columns, reduced within 16-lane group
  #pragma unroll
  for (int r = 0; r < 8; ++r) {
    float m = acc[0][r];
    #pragma unroll
    for (int j = 1; j < 8; ++j) m = fmaxf(m, acc[j][r]);
    #pragma unroll
    for (int o = 1; o < 16; o <<= 1) m = fmaxf(m, __shfl_xor(m, o, 32));
    if (ncol == 0) red_mx[r + half * 8][wave] = m;
  }
  __syncthreads();

  float rowmax[8];
  #pragma unroll
  for (int r = 0; r < 8; ++r) {
    float m = red_mx[r + half * 8][0];
    #pragma unroll
    for (int w = 1; w < 8; ++w) m = fmaxf(m, red_mx[r + half * 8][w]);
    rowmax[r] = m;
  }

  // exponentiate in place, per-row partial sums
  #pragma unroll
  for (int r = 0; r < 8; ++r) {
    float s = 0.0f;
    #pragma unroll
    for (int j = 0; j < 8; ++j) {
      float e = __expf(acc[j][r] - rowmax[r]);
      acc[j][r] = e;
      s += e;
    }
    #pragma unroll
    for (int o = 1; o < 16; o <<= 1) s += __shfl_xor(s, o, 32);
    if (ncol == 0) red_sm[r + half * 8][wave] = s;
  }
  __syncthreads();

  float inv[8];
  #pragma unroll
  for (int r = 0; r < 8; ++r) {
    float s = 0.0f;
    #pragma unroll
    for (int w = 0; w < 8; ++w) s += red_sm[r + half * 8][w];
    inv[r] = 1.0f / s;
  }

  unsigned short* dst = attn + (size_t)bat * SS * SS;
  #pragma unroll
  for (int j = 0; j < 8; ++j) {
    int n = nbase + j * 16 + ncol;
    #pragma unroll
    for (int r = 0; r < 8; ++r) {
      int m = mbase + r + half * 8;
      dst[(size_t)m * SS + n] = f2bf(acc[j][r] * inv[r]);
    }
  }
}

// --------------------------------------------------- ctx = attn @ h ------
__global__ __launch_bounds__(256) void gemm_ctx_kernel(
    const unsigned short* __restrict__ attn, const unsigned short* __restrict__ hT,
    unsigned short* __restrict__ ctx) {
  int lane  = threadIdx.x & 31;
  int wave  = threadIdx.x >> 5;
  int bat   = blockIdx.y;
  int mbase = blockIdx.x * 16;
  int nbase = wave * 64;
  const unsigned short* A  = attn + (size_t)bat * SS * SS;   // [S,S]
  const unsigned short* BT = hT   + (size_t)bat * HHH * SS;  // [H,S]
  int arow = mbase + (lane & 15);
  int kb   = (lane >> 4) << 3;
  v8f acc[4] = {};
  for (int k0 = 0; k0 < SS; k0 += 32) {
    v16bf a = load_frag(A + (size_t)arow * SS + k0 + kb);
    #pragma unroll
    for (int j = 0; j < 4; ++j) {
      int brow = nbase + j * 16 + (lane & 15);
      v16bf b = load_frag(BT + (size_t)brow * SS + k0 + kb);
      acc[j] = __builtin_amdgcn_wmma_f32_16x16x32_bf16(
          false, a, false, b, (short)0, acc[j], false, false);
    }
  }
  int half = lane >> 4;
  #pragma unroll
  for (int j = 0; j < 4; ++j) {
    int n = nbase + j * 16 + (lane & 15);
    #pragma unroll
    for (int r = 0; r < 8; ++r) {
      int m = mbase + r + half * 8;
      ctx[((size_t)bat * SS + m) * HHH + n] = f2bf(acc[j][r]);
    }
  }
}

// -------------------------- gate = sigmoid([h;ctx] @ Wg + bg); out = g*h --
__global__ __launch_bounds__(256) void gemm_gate_kernel(
    const unsigned short* __restrict__ hb, const unsigned short* __restrict__ ctx,
    const unsigned short* __restrict__ wgT, const float* __restrict__ bg,
    const float* __restrict__ h, float* __restrict__ out) {
  int lane  = threadIdx.x & 31;
  int wave  = threadIdx.x >> 5;
  int mbase = blockIdx.x * 16;
  int nbase = wave * 64;
  int arow  = mbase + (lane & 15);
  int kb    = (lane >> 4) << 3;
  v8f acc[4] = {};
  for (int k0 = 0; k0 < 2 * HHH; k0 += 32) {
    const unsigned short* ap = (k0 < HHH)
        ? hb  + (size_t)arow * HHH + k0 + kb
        : ctx + (size_t)arow * HHH + (k0 - HHH) + kb;
    v16bf a = load_frag(ap);
    #pragma unroll
    for (int j = 0; j < 4; ++j) {
      int brow = nbase + j * 16 + (lane & 15);
      v16bf b = load_frag(wgT + (size_t)brow * 1024 + k0 + kb);
      acc[j] = __builtin_amdgcn_wmma_f32_16x16x32_bf16(
          false, a, false, b, (short)0, acc[j], false, false);
    }
  }
  int half = lane >> 4;
  #pragma unroll
  for (int j = 0; j < 4; ++j) {
    int n  = nbase + j * 16 + (lane & 15);
    float bn = bg[n];
    #pragma unroll
    for (int r = 0; r < 8; ++r) {
      int m = mbase + r + half * 8;
      size_t idx = (size_t)m * HHH + n;
      float g = 1.0f / (1.0f + __expf(-(acc[j][r] + bn)));
      out[idx] = g * h[idx];
    }
  }
}

// ----------------------------------------------------------- avg ---------
__global__ __launch_bounds__(256) void avg_kernel(
    const float* __restrict__ cs, float* __restrict__ out_avg) {
  int idx = blockIdx.x * 256 + threadIdx.x;     // 0 .. B*H-1
  int b = idx >> 9, hc = idx & 511;
  float s = 0.0f;
  for (int t = 0; t < SS; ++t)
    s += cs[((size_t)(b * SS + t)) * HHH + hc];
  float m = s * (1.0f / (float)SS);
  #pragma unroll
  for (int k = 0; k < 4; ++k) out_avg[(size_t)k * BB * HHH + idx] = m;
}

// ----------------------------------------------------------------- host --
extern "C" void kernel_launch(void* const* d_in, const int* in_sizes, int n_in,
                              void* d_out, int out_size, void* d_ws, size_t ws_size,
                              hipStream_t stream) {
  (void)in_sizes; (void)n_in; (void)out_size; (void)ws_size;
  const int*   types = (const int*)d_in[0];
  const int*   ents  = (const int*)d_in[1];
  const int*   vals  = (const int*)d_in[2];
  const int*   ha    = (const int*)d_in[3];
  const float* Etp   = (const float*)d_in[4];
  const float* Eent  = (const float*)d_in[5];
  const float* Eval  = (const float*)d_in[6];
  const float* Eha   = (const float*)d_in[7];
  const float* Wenc  = (const float*)d_in[8];
  const float* benc  = (const float*)d_in[9];
  const float* Wg    = (const float*)d_in[10];
  const float* bg    = (const float*)d_in[11];
  float* out = (float*)d_out;

  char* ws = (char*)d_ws;
  size_t off = 0;
  auto carve = [&](size_t bytes) { char* p = ws + off; off += (bytes + 255) & ~(size_t)255; return p; };
  unsigned short* emb   = (unsigned short*)carve((size_t)MMM * EMBD * 2);
  unsigned short* wencT = (unsigned short*)carve((size_t)HHH * EMBD * 2);
  unsigned short* wgT   = (unsigned short*)carve((size_t)HHH * 1024 * 2);
  float*          h     = (float*)         carve((size_t)MMM * HHH * 4);
  unsigned short* hb    = (unsigned short*)carve((size_t)MMM * HHH * 2);
  unsigned short* hT    = (unsigned short*)carve((size_t)BB * HHH * SS * 2);
  unsigned short* attn  = (unsigned short*)carve((size_t)BB * SS * SS * 2);
  unsigned short* ctx   = (unsigned short*)carve((size_t)MMM * HHH * 2);

  embed_gather_kernel<<<MMM, 256, 0, stream>>>(types, ents, vals, ha,
                                               Etp, Eent, Eval, Eha, emb);
  transpose_wenc_kernel<<<(512 * 512) / 256, 256, 0, stream>>>(Wenc, wencT);
  transpose_wg_kernel<<<(512 * 1024) / 256, 256, 0, stream>>>(Wg, wgT);

  gemm_encode_kernel<<<MMM / 16, 256, 0, stream>>>(emb, wencT, benc, h, hb, hT);
  scores_softmax_kernel<<<dim3(SS / 16, BB), 256, 0, stream>>>(hb, attn);
  gemm_ctx_kernel<<<dim3(SS / 16, BB), 256, 0, stream>>>(attn, hT, ctx);
  gemm_gate_kernel<<<MMM / 16, 256, 0, stream>>>(hb, ctx, wgT, bg, h, out);
  avg_kernel<<<(BB * HHH) / 256, 256, 0, stream>>>(out, out + (size_t)MMM * HHH);
}
